// AutoformerModel_5334349381751
// MI455X (gfx1250) — matrix operand order repaired and checked by reading, not compiled, not run on hardware
//
#include <hip/hip_runtime.h>
#include <hip/hip_bf16.h>
#include <math.h>

// ---------------------------------------------------------------------------
// Autoformer forward for MI455X (gfx1250, wave32, WMMA).
// GEMM operands are staged once into f16 (weights converted once per launch,
// activations converted once per use or produced directly in f16), so the
// WMMA inner loop is b128 A loads + strided 16-bit B loads + v_wmma_f32_16x16x32_f16 with no
// per-iteration cvt. Each wave owns a 32x64 tile (2x4 WMMA tiles -> 8 WMMAs
// per k-step, ~21 FLOP/B intensity). The FFT autocorrelation is computed as
// S = Q K^T (f16 transpose + NN WMMA GEMM) + circular diagonal reduction.
// ---------------------------------------------------------------------------

typedef __attribute__((ext_vector_type(16))) _Float16 v16h;
typedef __attribute__((ext_vector_type(8)))  _Float16 v8h;
typedef __attribute__((ext_vector_type(8)))  float    v8f;

#define BB      32
#define LL      512
#define DMODEL  512
#define DFF     2048
#define CIN     7
#define DMARK   4
#define TOPK    18
#define PREDLEN 256
#define MROWS   (BB * LL)      // 16384 token rows (enc and dec both L=512)

static __host__ __device__ inline long long cdivll(long long a, long long b) { return (a + b - 1) / b; }

// ---------------------------------------------------------------------------
// WMMA GEMM: C[M,N] = A[M,K] @ B[K,N] (+bias) (+addend). A,B are f16
// row-major; C is f32 (outF16==0) or f16 (outF16==1); bias/addend f32.
// Batched via blockIdx.z (element strides sA/sB/sC; addend uses sC).
// Each wave computes a 32x64 tile (2x4 WMMA tiles); 4 waves per block
// stacked along M (block covers 128 rows x 64 cols).
// Requires M % 32 == 0, N % 64 == 0, K % 32 == 0.
// Fragment layouts per CDNA5 ISA 7.12.2 (wave32).
// ---------------------------------------------------------------------------
__global__ void gemm_wmma_kernel(const _Float16* __restrict__ A,
                                 const _Float16* __restrict__ Bm,
                                 const float* __restrict__ bias,
                                 const float* __restrict__ addend,
                                 void* __restrict__ Cout, int outF16,
                                 int M, int N, int K,
                                 long long sA, long long sB, long long sC)
{
    const int wave = threadIdx.x >> 5;
    const int lane = threadIdx.x & 31;
    const int n0   = blockIdx.x * 64;
    const int m0   = (blockIdx.y * 4 + wave) * 32;
    if (m0 >= M) return;                         // wave-uniform: EXEC stays all-1s

    const int batch = blockIdx.z;
    A  += (long long)batch * sA;
    Bm += (long long)batch * sB;

    const int mrow  = lane & 15;
    const int khalf = (lane >> 4) << 3;          // 0 for lanes 0-15, 8 for 16-31

    v8f acc[2][4] = {};
    for (int k0 = 0; k0 < K; k0 += 32) {
        if (k0 + 32 < K) {                       // prefetch next k-step panels
            __builtin_prefetch(A + (long long)(m0 + mrow) * K + (k0 + 32 + khalf), 0, 0);
            __builtin_prefetch(Bm + (long long)(k0 + 32 + lane) * N + n0, 0, 0);
        }
        v16h a[2], b[4];
        // A fragments (16x32 f16): per lane two contiguous 8-half runs
        #pragma unroll
        for (int mi = 0; mi < 2; ++mi) {
            const _Float16* Ar = A + (long long)(m0 + mi * 16 + mrow) * K + k0 + khalf;
            v8h lo = *(const v8h*)(Ar);
            v8h hi = *(const v8h*)(Ar + 16);
            #pragma unroll
            for (int j = 0; j < 8; ++j) { a[mi][j] = lo[j]; a[mi][j + 8] = hi[j]; }
        }
        // B fragments (32x16 f16): lane = n column, k rows khalf..+7 and 16+khalf..+7
        const _Float16* Br = Bm + (long long)(k0 + khalf) * N + n0 + mrow;
        #pragma unroll
        for (int ni = 0; ni < 4; ++ni) {
            #pragma unroll
            for (int j = 0; j < 8; ++j) { b[ni][j] = Br[(long long)j * N + ni * 16]; b[ni][j + 8] = Br[(long long)(j + 16) * N + ni * 16]; }
        }
        #pragma unroll
        for (int mi = 0; mi < 2; ++mi)
            #pragma unroll
            for (int ni = 0; ni < 4; ++ni)
                acc[mi][ni] = __builtin_amdgcn_wmma_f32_16x16x32_f16(
                    false, a[mi], false, b[ni], (short)0, acc[mi][ni], false, false);
    }

    // C/D layout: vgpr r: lanes 0-15 -> M=r, lanes 16-31 -> M=8+r; N = lane&15
    const int rbase = (lane >> 4) << 3;
    float*     C32 = (float*)Cout + (long long)batch * sC;
    _Float16*  C16 = (_Float16*)Cout + (long long)batch * sC;
    if (addend) addend += (long long)batch * sC;
    #pragma unroll
    for (int ni = 0; ni < 4; ++ni) {
        const int col = n0 + ni * 16 + mrow;
        const float bv = bias ? bias[col] : 0.f;
        #pragma unroll
        for (int mi = 0; mi < 2; ++mi) {
            #pragma unroll
            for (int r = 0; r < 8; ++r) {
                const int row = m0 + mi * 16 + rbase + r;
                float v = acc[mi][ni][r] + bv;
                if (addend) v += addend[(long long)row * N + col];
                if (outF16) C16[(long long)row * N + col] = (_Float16)v;
                else        C32[(long long)row * N + col] = v;
            }
        }
    }
}

// f32 -> f16 conversion, 4 elements per thread (n % 4 == 0)
__global__ void cvt16_kernel(const float* __restrict__ in, _Float16* __restrict__ out,
                             long long n)
{
    long long i = ((long long)blockIdx.x * blockDim.x + threadIdx.x) * 4;
    if (i >= n) return;
    float4 v = *(const float4*)(in + i);
    out[i + 0] = (_Float16)v.x;
    out[i + 1] = (_Float16)v.y;
    out[i + 2] = (_Float16)v.z;
    out[i + 3] = (_Float16)v.w;
}

// Batched LDS-tiled f16 transpose: out[b, c, r] = in[b, r, c].  R,C % 32 == 0.
__global__ void transpose16_kernel(const _Float16* __restrict__ in,
                                   _Float16* __restrict__ out, int R, int Cc)
{
    __shared__ _Float16 tile[32][33];
    const int b  = blockIdx.z;
    const int r0 = blockIdx.y * 32;
    const int c0 = blockIdx.x * 32;
    const _Float16* ib = in  + (long long)b * R * Cc;
    _Float16*       ob = out + (long long)b * R * Cc;
    const int tx = threadIdx.x & 31;
    const int ty = threadIdx.x >> 5;             // 256 threads: 32 x 8
    for (int i = ty; i < 32; i += 8)
        tile[i][tx] = ib[(long long)(r0 + i) * Cc + (c0 + tx)];
    __syncthreads();
    for (int i = ty; i < 32; i += 8)
        ob[(long long)(c0 + i) * R + (r0 + tx)] = tile[tx][i];
}

// ---------------------------------------------------------------------------
// Embedding: circular conv(width 3) over time on tokens + time-mark linear.
// ---------------------------------------------------------------------------
__global__ void embed_kernel(const float* __restrict__ x, const float* __restrict__ xm,
                             const float* __restrict__ tokw, const float* __restrict__ timw,
                             float* __restrict__ out, int L, int Cin, int Dm, int Mk)
{
    long long idx = (long long)blockIdx.x * blockDim.x + threadIdx.x;
    long long total = (long long)BB * L * Dm;
    if (idx >= total) return;
    int d = (int)(idx % Dm);
    int t = (int)((idx / Dm) % L);
    int b = (int)(idx / ((long long)Dm * L));
    float s = 0.f;
    for (int w = 0; w < 3; ++w) {
        int tt = t - 1 + w; tt = (tt + L) % L;   // circular pad
        const float* xr = x + ((long long)b * L + tt) * Cin;
        const float* wr = tokw + ((long long)w * Cin) * Dm + d;
        for (int c = 0; c < Cin; ++c) s += xr[c] * wr[(long long)c * Dm];
    }
    const float* mr = xm + ((long long)b * L + t) * Mk;
    for (int m = 0; m < Mk; ++m) s += mr[m] * timw[(long long)m * Dm + d];
    out[idx] = s;
}

// ---------------------------------------------------------------------------
// Series decomposition: 25-tap edge-replicated moving average.
// seas = x - ma; trend written (= or +=) if non-null.
// ---------------------------------------------------------------------------
__global__ void decomp_kernel(const float* __restrict__ x,
                              float* __restrict__ seas, float* __restrict__ trend,
                              int L, int D, int accumTrend)
{
    long long idx = (long long)blockIdx.x * blockDim.x + threadIdx.x;
    long long total = (long long)BB * L * D;
    if (idx >= total) return;
    int d = (int)(idx % D);
    int t = (int)((idx / D) % L);
    int b = (int)(idx / ((long long)D * L));
    float s = 0.f;
    for (int j = -12; j <= 12; ++j) {
        int tt = t + j; tt = tt < 0 ? 0 : (tt > L - 1 ? L - 1 : tt);
        s += x[((long long)b * L + tt) * D + d];
    }
    s *= (1.f / 25.f);
    if (trend) { if (accumTrend) trend[idx] += s; else trend[idx] = s; }
    if (seas) seas[idx] = x[idx] - s;
}

// mean over time axis: out[b,d] = mean_t x[b,t,d]
__global__ void colmean_kernel(const float* __restrict__ x, float* __restrict__ out,
                               int L, int D)
{
    int idx = blockIdx.x * blockDim.x + threadIdx.x;
    if (idx >= BB * D) return;
    int d = idx % D, b = idx / D;
    float s = 0.f;
    for (int t = 0; t < L; ++t) s += x[((long long)b * L + t) * D + d];
    out[idx] = s / (float)L;
}

__global__ void sub_colmean_kernel(float* __restrict__ x, const float* __restrict__ mean,
                                   int L, int D)
{
    long long idx = (long long)blockIdx.x * blockDim.x + threadIdx.x;
    long long total = (long long)BB * L * D;
    if (idx >= total) return;
    int d = (int)(idx % D);
    int b = (int)(idx / ((long long)D * L));
    x[idx] -= mean[b * D + d];
}

// decoder init: label part from enc decomp tail, pred part zeros / enc-mean
__global__ void dec_init_kernel(const float* __restrict__ seasE, const float* __restrict__ trendE,
                                const float* __restrict__ meanE,
                                float* __restrict__ seasI, float* __restrict__ trendI)
{
    int idx = blockIdx.x * blockDim.x + threadIdx.x;
    if (idx >= BB * LL * CIN) return;
    int c = idx % CIN;
    int t = (idx / CIN) % LL;
    int b = idx / (CIN * LL);
    if (t < PREDLEN) {  // LABEL_LEN == 256: rows 256..511 of encoder decomp
        long long src = ((long long)b * LL + (PREDLEN + t)) * CIN + c;
        seasI[idx]  = seasE[src];
        trendI[idx] = trendE[src];
    } else {
        seasI[idx]  = 0.f;
        trendI[idx] = meanE[b * CIN + c];
    }
}

// LayerNorm over channels (one 256-thread block per row)
__global__ void ln_kernel(const float* __restrict__ x, const float* __restrict__ w,
                          const float* __restrict__ b, float* __restrict__ out, int D)
{
    __shared__ float red[256];
    const int row = blockIdx.x;
    const float* xr = x + (long long)row * D;
    float s = 0.f;
    for (int i = threadIdx.x; i < D; i += 256) s += xr[i];
    red[threadIdx.x] = s; __syncthreads();
    for (int st = 128; st > 0; st >>= 1) {
        if (threadIdx.x < st) red[threadIdx.x] += red[threadIdx.x + st];
        __syncthreads();
    }
    const float mu = red[0] / (float)D;
    __syncthreads();
    float v = 0.f;
    for (int i = threadIdx.x; i < D; i += 256) { float dd = xr[i] - mu; v += dd * dd; }
    red[threadIdx.x] = v; __syncthreads();
    for (int st = 128; st > 0; st >>= 1) {
        if (threadIdx.x < st) red[threadIdx.x] += red[threadIdx.x + st];
        __syncthreads();
    }
    const float rstd = rsqrtf(red[0] / (float)D + 1e-5f);
    for (int i = threadIdx.x; i < D; i += 256)
        out[(long long)row * D + i] = (xr[i] - mu) * rstd * w[i] + b[i];
}

// exact GELU, f32 in -> f16 out (only ever consumed by a GEMM)
__global__ void gelu16_kernel(const float* __restrict__ in, _Float16* __restrict__ out,
                              long long n)
{
    long long idx = (long long)blockIdx.x * blockDim.x + threadIdx.x;
    if (idx >= n) return;
    float v = in[idx];
    out[idx] = (_Float16)(0.5f * v * (1.f + erff(v * 0.70710678118654752f)));
}

// corr[b,tau] = (1/D) * sum_t S[b, t, (t - tau) mod L]   (circular autocorr)
__global__ void corr_mean_kernel(const float* __restrict__ S, float* __restrict__ corr,
                                 int L, float invD)
{
    int idx = blockIdx.x * blockDim.x + threadIdx.x;
    if (idx >= BB * L) return;
    int tau = idx % L, b = idx / L;
    const float* Sb = S + (long long)b * L * L;
    float s = 0.f;
    for (int t = 0; t < L; ++t) {
        int s2 = t - tau; if (s2 < 0) s2 += L;
        s += Sb[(long long)t * L + s2];
    }
    corr[idx] = s * invD;
}

// per-batch top-18 delays + softmax weights
__global__ void topk_kernel(const float* __restrict__ corr, float* __restrict__ wts,
                            int* __restrict__ dls, int L)
{
    __shared__ float buf[LL];
    const int b = blockIdx.x;
    for (int i = threadIdx.x; i < L; i += blockDim.x) buf[i] = corr[b * L + i];
    __syncthreads();
    if (threadIdx.x == 0) {
        float w[TOPK]; int dd[TOPK];
        for (int i = 0; i < TOPK; ++i) {
            int best = 0; float bv = -3.4e38f;
            for (int j = 0; j < L; ++j)
                if (buf[j] > bv) { bv = buf[j]; best = j; }
            w[i] = bv; dd[i] = best; buf[best] = -3.4e38f;
        }
        const float mx = w[0];
        float se = 0.f;
        for (int i = 0; i < TOPK; ++i) { w[i] = expf(w[i] - mx); se += w[i]; }
        const float inv = 1.f / se;
        for (int i = 0; i < TOPK; ++i) { wts[b * TOPK + i] = w[i] * inv; dls[b * TOPK + i] = dd[i]; }
    }
}

// out[b,t,d] = sum_i w[b,i] * v[b,(t+delay[b,i]) mod L, d]   (f16 in/out, f32 math)
__global__ void agg_kernel(const _Float16* __restrict__ v, const float* __restrict__ wts,
                           const int* __restrict__ dls, _Float16* __restrict__ out,
                           int L, int D)
{
    long long idx = (long long)blockIdx.x * blockDim.x + threadIdx.x;
    long long total = (long long)BB * L * D;
    if (idx >= total) return;
    int d = (int)(idx % D);
    int t = (int)((idx / D) % L);
    int b = (int)(idx / ((long long)D * L));
    float s = 0.f;
    for (int i = 0; i < TOPK; ++i) {
        int tt = t + dls[b * TOPK + i]; if (tt >= L) tt -= L;
        s += wts[b * TOPK + i] * (float)v[((long long)b * L + tt) * D + d];
    }
    out[idx] = (_Float16)s;
}

// trend[b,t,c] += circular_conv3(tin, wt)  with wt (3, D, C)
__global__ void trend_conv_kernel(const float* __restrict__ tin, const float* __restrict__ wt,
                                  float* __restrict__ trend, int L, int D, int C)
{
    int idx = blockIdx.x * blockDim.x + threadIdx.x;
    if (idx >= BB * L * C) return;
    int c = idx % C;
    int t = (idx / C) % L;
    int b = idx / (C * L);
    float s = 0.f;
    for (int w = 0; w < 3; ++w) {
        int tt = (t - 1 + w + L) % L;
        const float* xr = tin + ((long long)b * L + tt) * D;
        const float* wr = wt + ((long long)w * D) * C + c;
        for (int d = 0; d < D; ++d) s += xr[d] * wr[(long long)d * C];
    }
    trend[idx] += s;
}

// out[b,t,c] = (x[b,256+t,:] @ proj) + pb + trend[b,256+t,c]
__global__ void final_kernel(const float* __restrict__ x, const float* __restrict__ pw,
                             const float* __restrict__ pb, const float* __restrict__ trend,
                             float* __restrict__ out, int L, int D, int C)
{
    int idx = blockIdx.x * blockDim.x + threadIdx.x;
    if (idx >= BB * PREDLEN * C) return;
    int c = idx % C;
    int t = (idx / C) % PREDLEN;
    int b = idx / (C * PREDLEN);
    int tt = t + (L - PREDLEN);
    const float* xr = x + ((long long)b * L + tt) * D;
    float s = pb[c];
    for (int d = 0; d < D; ++d) s += xr[d] * pw[d * C + c];
    out[idx] = s + trend[((long long)b * L + tt) * C + c];
}

// ---------------------------------------------------------------------------
// Host orchestration
// ---------------------------------------------------------------------------
struct AttnP  { const float *bk, *bo, *bq, *bv, *wk, *wo, *wq, *wv; };
struct AttnW16 { _Float16 *wq, *wk, *wv, *wo; };

extern "C" void kernel_launch(void* const* d_in, const int* in_sizes, int n_in,
                              void* d_out, int out_size, void* d_ws, size_t ws_size,
                              hipStream_t stream)
{
    (void)in_sizes; (void)n_in; (void)out_size;

    // ---- inputs (top level in insertion order, params in sorted pytree order)
    const float* x_enc  = (const float*)d_in[0];   // (32,512,7)
    const float* xm_enc = (const float*)d_in[1];   // (32,512,4)
    /* d_in[2] = x_dec: values unused by the reference */
    const float* xm_dec = (const float*)d_in[3];   // (32,512,4)

    int ii = 4;
    auto P = [&](void) { return (const float*)d_in[ii++]; };
    const float* dec_time_w = P();                 // dec_emb.time_w (4,512)
    const float* dec_tok_w  = P();                 // dec_emb.token_w (3,7,512)
    const float* d_conv1 = P();                    // dec conv1 (512,2048)
    const float* d_conv2 = P();                    // dec conv2 (2048,512)
    AttnP cross; cross.bk = P(); cross.bo = P(); cross.bq = P(); cross.bv = P();
                 cross.wk = P(); cross.wo = P(); cross.wq = P(); cross.wv = P();
    AttnP dself; dself.bk = P(); dself.bo = P(); dself.bq = P(); dself.bv = P();
                 dself.wk = P(); dself.wo = P(); dself.wq = P(); dself.wv = P();
    const float* trend_proj = P();                 // (3,512,7)
    const float* dec_norm_b = P();
    const float* dec_norm_w = P();
    const float* dec_proj_b = P();                 // (7)
    const float* dec_proj_w = P();                 // (512,7)
    const float* enc_time_w = P();
    const float* enc_tok_w  = P();
    AttnP eattn[2]; const float* e_conv1[2]; const float* e_conv2[2];
    for (int l = 0; l < 2; ++l) {
        eattn[l].bk = P(); eattn[l].bo = P(); eattn[l].bq = P(); eattn[l].bv = P();
        eattn[l].wk = P(); eattn[l].wo = P(); eattn[l].wq = P(); eattn[l].wv = P();
        e_conv1[l] = P(); e_conv2[l] = P();
    }
    const float* enc_norm_b = P();
    const float* enc_norm_w = P();

    // ---- workspace layout (256B-aligned byte allocator)
    const size_t MD  = (size_t)MROWS * DMODEL;     // 8,388,608 elements
    const size_t MF  = (size_t)MROWS * DFF;        // 33,554,432 elements
    char* base = (char*)d_ws;
    size_t off = 0;
    auto takeB = [&](size_t bytes) { void* p = base + off; off += (bytes + 255) & ~(size_t)255; return p; };

    // f32 buffers
    float* bx    = (float*)takeB(MD * 4);          // current activation
    float* by    = (float*)takeB(MD * 4);          // residual/temp
    float* bs    = (float*)takeB(MD * 4);          // scores (B,L,L)
    float* benc  = (float*)takeB(MD * 4);          // encoder output
    float* bt    = (float*)takeB(MD * 4);          // decoder trend accumulator
    float* bff   = (float*)takeB(MF * 4);          // FFN mid (pre-GELU)
    float* bcorr = (float*)takeB((size_t)BB * LL * 4);
    float* btw   = (float*)takeB((size_t)BB * TOPK * 4);
    int*   btd   = (int*)  takeB((size_t)BB * TOPK * 4);
    float* bseas = (float*)takeB((size_t)BB * LL * CIN * 4);
    float* btrd  = (float*)takeB((size_t)BB * LL * CIN * 4);
    float* bse   = (float*)takeB((size_t)BB * LL * CIN * 4);
    float* bte   = (float*)takeB((size_t)BB * LL * CIN * 4);
    float* bmean = (float*)takeB((size_t)BB * CIN * 4);
    float* btm   = (float*)takeB((size_t)BB * DMODEL * 4);

    // f16 buffers
    _Float16* h_a  = (_Float16*)takeB(MD * 2);     // activation (A operand)
    _Float16* h_b  = (_Float16*)takeB(MD * 2);     // second activation (cross-attn kv)
    _Float16* h_q  = (_Float16*)takeB(MD * 2);
    _Float16* h_k  = (_Float16*)takeB(MD * 2);     // K; later reused as agg output
    _Float16* h_v  = (_Float16*)takeB(MD * 2);
    _Float16* h_kt = (_Float16*)takeB(MD * 2);     // K^T
    _Float16* h_ff = (_Float16*)takeB(MF * 2);     // post-GELU
    const size_t WATT = (size_t)DMODEL * DMODEL;   // 262,144
    const size_t WFF  = (size_t)DMODEL * DFF;      // 1,048,576
    _Float16* h_w = (_Float16*)takeB((4 * 4 * WATT + 3 * 2 * WFF) * 2);

    if (off > ws_size) return;                     // deterministic no-op if too small

    auto ewg = [](long long n) { return dim3((unsigned)cdivll(n, 256)); };
    auto cvt = [&](const float* in, _Float16* out, long long n) {
        cvt16_kernel<<<ewg(n / 4), 256, 0, stream>>>(in, out, n);
    };

    // convert all GEMM weights to f16 once
    size_t wo = 0;
    auto wtake = [&](const float* src, size_t n) {
        _Float16* p = h_w + wo; wo += n; cvt(src, p, (long long)n); return p;
    };
    AttnW16 w_e[2], w_s, w_c;
    _Float16 *c1_16[3], *c2_16[3];                 // enc0, enc1, dec
    for (int l = 0; l < 2; ++l) {
        w_e[l].wq = wtake(eattn[l].wq, WATT); w_e[l].wk = wtake(eattn[l].wk, WATT);
        w_e[l].wv = wtake(eattn[l].wv, WATT); w_e[l].wo = wtake(eattn[l].wo, WATT);
        c1_16[l] = wtake(e_conv1[l], WFF);    c2_16[l] = wtake(e_conv2[l], WFF);
    }
    w_s.wq = wtake(dself.wq, WATT); w_s.wk = wtake(dself.wk, WATT);
    w_s.wv = wtake(dself.wv, WATT); w_s.wo = wtake(dself.wo, WATT);
    w_c.wq = wtake(cross.wq, WATT); w_c.wk = wtake(cross.wk, WATT);
    w_c.wv = wtake(cross.wv, WATT); w_c.wo = wtake(cross.wo, WATT);
    c1_16[2] = wtake(d_conv1, WFF); c2_16[2] = wtake(d_conv2, WFF);

    auto gemm = [&](const _Float16* A, const _Float16* Bm, const float* bias,
                    const float* add, void* Cc, int outF16, int M, int N, int K,
                    long long sA, long long sB, long long sC, int batch) {
        dim3 g((unsigned)(N / 64), (unsigned)cdivll(M, 128), (unsigned)batch);
        gemm_wmma_kernel<<<g, 128, 0, stream>>>(A, Bm, bias, add, Cc, outF16, M, N, K, sA, sB, sC);
    };
    auto decomp = [&](const float* xin, float* seas, float* trend, int accum) {
        decomp_kernel<<<ewg((long long)MROWS * DMODEL), 256, 0, stream>>>(
            xin, seas, trend, LL, DMODEL, accum);
    };
    // autocorrelation attention: outbuf = xq + AC(q,k,v) @ wo + bo
    auto ac_attn = [&](const float* xq, const float* xkv, const AttnP& p, const AttnW16& w,
                       float* outbuf) {
        cvt(xq, h_a, (long long)MD);
        const _Float16* kv16 = h_a;
        if (xkv != xq) { cvt(xkv, h_b, (long long)MD); kv16 = h_b; }
        gemm(h_a,  w.wq, p.bq, nullptr, h_q, 1, MROWS, DMODEL, DMODEL, 0, 0, 0, 1);
        gemm(kv16, w.wk, p.bk, nullptr, h_k, 1, MROWS, DMODEL, DMODEL, 0, 0, 0, 1);
        gemm(kv16, w.wv, p.bv, nullptr, h_v, 1, MROWS, DMODEL, DMODEL, 0, 0, 0, 1);
        transpose16_kernel<<<dim3(DMODEL / 32, LL / 32, BB), 256, 0, stream>>>(h_k, h_kt, LL, DMODEL);
        // S = Q @ K^T per batch (f32 out)
        gemm(h_q, h_kt, nullptr, nullptr, bs, 0, LL, LL, DMODEL,
             (long long)LL * DMODEL, (long long)LL * DMODEL, (long long)LL * LL, BB);
        corr_mean_kernel<<<ewg(BB * LL), 256, 0, stream>>>(bs, bcorr, LL, 1.f / (float)DMODEL);
        topk_kernel<<<BB, 256, 0, stream>>>(bcorr, btw, btd, LL);
        // aggregation reuses h_k (K fully consumed by transpose)
        agg_kernel<<<ewg((long long)MROWS * DMODEL), 256, 0, stream>>>(h_v, btw, btd, h_k, LL, DMODEL);
        gemm(h_k, w.wo, p.bo, xq, outbuf, 0, MROWS, DMODEL, DMODEL, 0, 0, 0, 1);
    };
    auto ffn = [&](const float* xin, const _Float16* c1, const _Float16* c2, float* outbuf) {
        cvt(xin, h_a, (long long)MD);
        gemm(h_a, c1, nullptr, nullptr, bff, 0, MROWS, DFF, DMODEL, 0, 0, 0, 1);
        gelu16_kernel<<<ewg((long long)MF), 256, 0, stream>>>(bff, h_ff, (long long)MF);
        gemm(h_ff, c2, nullptr, xin, outbuf, 0, MROWS, DMODEL, DFF, 0, 0, 0, 1);
    };

    // ======================= encoder =======================
    embed_kernel<<<ewg((long long)MROWS * DMODEL), 256, 0, stream>>>(
        x_enc, xm_enc, enc_tok_w, enc_time_w, bx, LL, CIN, DMODEL, DMARK);

    for (int l = 0; l < 2; ++l) {
        ac_attn(bx, bx, eattn[l], w_e[l], by);               // by = x + attn
        decomp(by, bx, nullptr, 0);                          // bx = seasonal
        ffn(bx, c1_16[l], c2_16[l], by);                     // by = x + ffn
        decomp(by, bx, nullptr, 0);
    }
    // _my_layernorm on encoder output -> benc
    ln_kernel<<<MROWS, 256, 0, stream>>>(bx, enc_norm_w, enc_norm_b, benc, DMODEL);
    colmean_kernel<<<ewg(BB * DMODEL), 256, 0, stream>>>(benc, btm, LL, DMODEL);
    sub_colmean_kernel<<<ewg((long long)MROWS * DMODEL), 256, 0, stream>>>(benc, btm, LL, DMODEL);

    // ======================= decoder init =======================
    decomp_kernel<<<ewg((long long)BB * LL * CIN), 256, 0, stream>>>(x_enc, bse, bte, LL, CIN, 0);
    colmean_kernel<<<ewg(BB * CIN), 256, 0, stream>>>(x_enc, bmean, LL, CIN);
    dec_init_kernel<<<ewg(BB * LL * CIN), 256, 0, stream>>>(bse, bte, bmean, bseas, btrd);
    embed_kernel<<<ewg((long long)MROWS * DMODEL), 256, 0, stream>>>(
        bseas, xm_dec, dec_tok_w, dec_time_w, bx, LL, CIN, DMODEL, DMARK);

    // ======================= decoder layer =======================
    ac_attn(bx, bx, dself, w_s, by);                         // self attention
    decomp(by, bx, bt, 0);                                   // t1
    ac_attn(bx, benc, cross, w_c, by);                       // cross attention vs encoder
    decomp(by, bx, bt, 1);                                   // += t2
    ffn(bx, c1_16[2], c2_16[2], by);
    decomp(by, bx, bt, 1);                                   // += t3
    trend_conv_kernel<<<ewg(BB * LL * CIN), 256, 0, stream>>>(bt, trend_proj, btrd, LL, DMODEL, CIN);

    // ======================= head =======================
    ln_kernel<<<MROWS, 256, 0, stream>>>(bx, dec_norm_w, dec_norm_b, by, DMODEL);
    colmean_kernel<<<ewg(BB * DMODEL), 256, 0, stream>>>(by, btm, LL, DMODEL);
    sub_colmean_kernel<<<ewg((long long)MROWS * DMODEL), 256, 0, stream>>>(by, btm, LL, DMODEL);
    final_kernel<<<ewg(BB * PREDLEN * CIN), 256, 0, stream>>>(
        by, dec_proj_w, dec_proj_b, btrd, (float*)d_out, LL, DMODEL, CIN);
}
